// MultiHeadAttention_10222022165088
// MI455X (gfx1250) — matrix operand order repaired and checked by reading, not compiled, not run on hardware
//
#include <hip/hip_runtime.h>
#include <hip/hip_bf16.h>

// ---------------------------------------------------------------------------
// Multi-head attention forward for MI455X (gfx1250, wave32, WMMA, async-LDS)
//   B=2, S=2048, H=16, Dk=Dv=64, D=1024
//   d_out = [out: 2*2048*1024 f32][attn: 2*16*2048*2048 f32]
// ---------------------------------------------------------------------------

typedef __attribute__((ext_vector_type(16))) _Float16 v16h;
typedef __attribute__((ext_vector_type(8)))  _Float16 v8h;
typedef __attribute__((ext_vector_type(4)))  _Float16 v4h;
typedef __attribute__((ext_vector_type(8)))  float    v8f;

#define BATCH   2
#define SEQ     2048
#define NHEAD   16
#define DMODEL  1024
#define DK      64
#define SSTR    2052            // f32 LDS score-row stride (16B-aligned rows, bank-staggered)
#define LDA_STR 40              // f16 LDS A-tile stride (16B-aligned rows, conflict-free frags)
#define LDB_STR 56              // f16 LDS B-tile stride
#define LDQ_STR 72              // f16 LDS Q-tile stride

// ---- CDNA5 async global->LDS copy (raw bytes, ASYNCcnt-tracked) ----------
static __device__ __forceinline__ void async_b128_to_lds(uint32_t lds_byte_off,
                                                         const void* gptr) {
    asm volatile("global_load_async_to_lds_b128 %0, %1, off"
                 :: "v"(lds_byte_off), "v"(gptr) : "memory");
}
static __device__ __forceinline__ void async_wait0() {
#if __has_builtin(__builtin_amdgcn_s_wait_asynccnt)
    __builtin_amdgcn_s_wait_asynccnt(0);
#else
    asm volatile("s_wait_asynccnt 0x0" ::: "memory");
#endif
}

static __device__ __forceinline__ v8f wmma_f16(v16h a, v16h b, v8f c) {
    return __builtin_amdgcn_wmma_f32_16x16x32_f16(
        false, a, false, b, (short)0, c, false, false);
}

// A fragment (16x32 f16): lane l: row (l&15), k = kbase+(l>>4)*8 + {0..7, 16..23}
static __device__ __forceinline__ v16h load_a_frag(const _Float16* base, int stride,
                                                   int row, int khalf) {
    union { v16h v; v8h h[2]; } u;
    const _Float16* p = base + row * stride + khalf;
    u.h[0] = *(const v8h*)(p);
    u.h[1] = *(const v8h*)(p + 16);
    return u.v;
}

// B fragment (32x16 f16) from [N][K]-layout buffer: lane l: col (l&15),
// k = (l>>4)*8 + {0..7, 16..23}
static __device__ __forceinline__ v16h load_b_frag_lds(const _Float16* base, int stride,
                                                       int col, int kb2) {
    union { v16h v; v8h h[2]; } u;
    const _Float16* p = base + col * stride + kb2;
    u.h[0] = *(const v8h*)(p);
    u.h[1] = *(const v8h*)(p + 16);
    return u.v;
}

// ---------------------------------------------------------------------------
// Kernel 1: QKV projections (z = 0/1/2 -> Q/K/V).
//   X[4096,1024]f32 @ W[1024,1024]f32 -> f16 head layouts:
//   Q,K: [b][h][s][64]   V: [b][h][64][s] (transposed for P*V B-fragments)
// Block tile 128x128, 8 waves in 4x2, each wave 32x64 = 8 WMMA / k-step.
// ---------------------------------------------------------------------------
__global__ __launch_bounds__(256) void qkv_proj_kernel(
    const float* __restrict__ xq, const float* __restrict__ xk, const float* __restrict__ xv,
    const float* __restrict__ wq, const float* __restrict__ wk, const float* __restrict__ wv,
    _Float16* __restrict__ Qh, _Float16* __restrict__ Kh, _Float16* __restrict__ Vt)
{
    const int mode = blockIdx.z;
    const float* __restrict__ X = (mode == 0) ? xq : (mode == 1) ? xk : xv;
    const float* __restrict__ W = (mode == 0) ? wq : (mode == 1) ? wk : wv;

    __shared__ _Float16 As[128 * LDA_STR];
    __shared__ _Float16 Bs[128 * LDB_STR];

    const int tid  = threadIdx.x;
    const int wave = tid >> 5;
    const int lane = tid & 31;
    const int M0 = blockIdx.y * 128;
    const int N0 = blockIdx.x * 128;
    const int wm = wave >> 1;        // 0..3
    const int wn = wave & 1;         // 0..1
    const int ln = lane & 15;
    const int kh  = (lane >> 4) * 8;
    const int kb2 = (lane >> 4) * 16;

    v8f acc[2][4] = {};

    for (int k0 = 0; k0 < DMODEL; k0 += 32) {
        // A tile 128x32: 1024 float4, 4 per thread, b128 loads + packed b64 stores
        #pragma unroll
        for (int i = 0; i < 4; ++i) {
            int idx4 = tid + i * 256;
            int row = idx4 >> 3, c4 = idx4 & 7;
            float4 f = *(const float4*)&X[(size_t)(M0 + row) * DMODEL + k0 + c4 * 4];
            v4h hv = { (_Float16)f.x, (_Float16)f.y, (_Float16)f.z, (_Float16)f.w };
            *(v4h*)&As[row * LDA_STR + c4 * 4] = hv;
        }
        // B tile 32x128 transposed -> Bs[n][k]: 1024 float4, 4 per thread
        #pragma unroll
        for (int i = 0; i < 4; ++i) {
            int idx4 = tid + i * 256;
            int kr = idx4 >> 5, c4 = idx4 & 31;
            float4 f = *(const float4*)&W[(size_t)(k0 + kr) * DMODEL + N0 + c4 * 4];
            Bs[(c4 * 4 + 0) * LDB_STR + kr] = (_Float16)f.x;
            Bs[(c4 * 4 + 1) * LDB_STR + kr] = (_Float16)f.y;
            Bs[(c4 * 4 + 2) * LDB_STR + kr] = (_Float16)f.z;
            Bs[(c4 * 4 + 3) * LDB_STR + kr] = (_Float16)f.w;
        }
        if (k0 + 32 < DMODEL)
            __builtin_prefetch(&X[(size_t)(M0 + (tid >> 3)) * DMODEL + k0 + 32], 0, 0);
        __syncthreads();

        v16h a0 = load_a_frag(As, LDA_STR, wm * 32 + ln,      kh);
        v16h a1 = load_a_frag(As, LDA_STR, wm * 32 + 16 + ln, kh);
        v16h bf[4];
        #pragma unroll
        for (int nt = 0; nt < 4; ++nt)
            bf[nt] = load_b_frag_lds(Bs, LDB_STR, wn * 64 + nt * 16 + ln, kh);
        #pragma unroll
        for (int nt = 0; nt < 4; ++nt) {
            acc[0][nt] = wmma_f16(a0, bf[nt], acc[0][nt]);
            acc[1][nt] = wmma_f16(a1, bf[nt], acc[1][nt]);
        }
        __syncthreads();
    }

    #pragma unroll
    for (int mt = 0; mt < 2; ++mt)
    #pragma unroll
    for (int nt = 0; nt < 4; ++nt)
    #pragma unroll
    for (int r = 0; r < 8; ++r) {
        int gm = M0 + wm * 32 + mt * 16 + r + ((lane >> 4) << 3);
        int gn = N0 + wn * 64 + nt * 16 + ln;
        _Float16 hv = (_Float16)acc[mt][nt][r];
        int b = gm >> 11, s = gm & (SEQ - 1);
        int hh = gn >> 6, d = gn & 63;
        if (mode == 0)
            Qh[((size_t)(b * NHEAD + hh) * SEQ + s) * DK + d] = hv;
        else if (mode == 1)
            Kh[((size_t)(b * NHEAD + hh) * SEQ + s) * DK + d] = hv;
        else
            Vt[((size_t)(b * NHEAD + hh) * DK + d) * SEQ + s] = hv;
    }
}

// ---------------------------------------------------------------------------
// Kernel 2: attention. One workgroup per (b, h, 32-row q tile).
//   262 KB f32 score tile resident in LDS (needs CDNA5's 320KB/WG).
//   Q tile staged with global_load_async_to_lds_b128 (ASYNCcnt path).
// ---------------------------------------------------------------------------
__global__ __launch_bounds__(256) void attn_kernel(
    const _Float16* __restrict__ Qh, const _Float16* __restrict__ Kh,
    const _Float16* __restrict__ Vt, _Float16* __restrict__ ctx,
    float* __restrict__ attn_out)
{
    extern __shared__ char smem[];
    _Float16* Qs = (_Float16*)smem;                       // 32 x LDQ_STR halves
    float*    sc = (float*)(smem + 32 * LDQ_STR * 2);     // 32 x SSTR floats

    const int tid  = threadIdx.x;
    const int wave = tid >> 5;
    const int lane = tid & 31;
    const int ln   = lane & 15;
    const int kh   = (lane >> 4) * 8;
    const int kb2  = (lane >> 4) * 16;

    const int q0 = blockIdx.x * 32;
    const int h  = blockIdx.y;
    const int b  = blockIdx.z;

    const _Float16* Qbase = Qh + ((size_t)(b * NHEAD + h) * SEQ + q0) * DK;
    const _Float16* Kbase = Kh + (size_t)(b * NHEAD + h) * SEQ * DK;
    const _Float16* Vbase = Vt + (size_t)(b * NHEAD + h) * DK * SEQ;

    // ---- async-copy 32x64 f16 Q tile into LDS (1 b128 per thread) -------
    {
        const uint32_t dynb = __builtin_amdgcn_groupstaticsize(); // dyn-LDS base
        int row  = tid >> 3;            // 8 threads per 64-half row
        int colh = (tid & 7) * 8;       // 8 halves = 16B per thread
        uint32_t loff = dynb + (uint32_t)(row * LDQ_STR + colh) * 2;
        async_b128_to_lds(loff, Qbase + row * DK + colh);
    }
    async_wait0();
    __syncthreads();

    // ---- Phase 1: scores = Q K^T * (1/sqrt(64)) -------------------------
    v16h aq[2][2];
    #pragma unroll
    for (int mt = 0; mt < 2; ++mt)
        #pragma unroll
        for (int ks = 0; ks < 2; ++ks)
            aq[mt][ks] = load_a_frag(Qs, LDQ_STR, mt * 16 + ln, ks * 32 + kh);

    for (int nt = wave; nt < SEQ / 16; nt += 8) {
        v8f s0 = {}, s1 = {};
        #pragma unroll
        for (int ks = 0; ks < 2; ++ks) {
            // K rows are directly valid B fragments (Bt[n][k] == Kh[s][k])
            v16h bk = load_b_frag_lds(Kbase, DK, nt * 16 + ln, ks * 32 + kh);
            s0 = wmma_f16(aq[0][ks], bk, s0);
            s1 = wmma_f16(aq[1][ks], bk, s1);
        }
        #pragma unroll
        for (int r = 0; r < 8; ++r) {
            int mr = r + ((lane >> 4) << 3);
            sc[mr * SSTR + nt * 16 + ln]        = s0[r] * 0.125f;
            sc[(16 + mr) * SSTR + nt * 16 + ln] = s1[r] * 0.125f;
        }
    }
    __syncthreads();

    // ---- Phase 2: softmax per row (float4 vectorized), stream attn ------
    float* arow_base = attn_out + ((size_t)(b * NHEAD + h) * SEQ + q0) * SEQ;
    #pragma unroll
    for (int rr = 0; rr < 4; ++rr) {
        int row = wave * 4 + rr;
        float* rowp = sc + row * SSTR;

        float mx = -3.0e38f;
        for (int c = lane * 4; c < SEQ; c += 128) {
            float4 f = *(const float4*)(rowp + c);
            mx = fmaxf(mx, fmaxf(fmaxf(f.x, f.y), fmaxf(f.z, f.w)));
        }
        #pragma unroll
        for (int off = 16; off > 0; off >>= 1) mx = fmaxf(mx, __shfl_xor(mx, off, 32));

        float sum = 0.0f;
        for (int c = lane * 4; c < SEQ; c += 128) {
            float4 f = *(const float4*)(rowp + c);
            f.x = __expf(f.x - mx); f.y = __expf(f.y - mx);
            f.z = __expf(f.z - mx); f.w = __expf(f.w - mx);
            sum += (f.x + f.y) + (f.z + f.w);
            *(float4*)(rowp + c) = f;
        }
        #pragma unroll
        for (int off = 16; off > 0; off >>= 1) sum += __shfl_xor(sum, off, 32);

        float inv = 1.0f / sum;
        float* arow = arow_base + (size_t)row * SEQ;
        for (int c = lane * 4; c < SEQ; c += 128) {
            float4 f = *(const float4*)(rowp + c);
            f.x *= inv; f.y *= inv; f.z *= inv; f.w *= inv;
            *(float4*)(rowp + c) = f;
            *(float4*)(arow + c) = f;          // b128 global store of attn
        }
    }
    __syncthreads();

    // ---- Phase 3: O = P V (M=32,N=64,K=2048), one 16x16 tile per wave ---
    const int mt = wave >> 2;   // 0..1
    const int nt = wave & 3;    // 0..3
    v8f o = {};
    for (int ks = 0; ks < SEQ; ks += 32) {
        const float4* ap = (const float4*)(sc + (mt * 16 + ln) * SSTR + ks + kh);
        float4 f0 = ap[0], f1 = ap[1], f2 = ap[4], f3 = ap[5];  // k:+0..7, +16..23
        union { v16h v; v4h q[4]; } ua;
        ua.q[0] = (v4h){ (_Float16)f0.x, (_Float16)f0.y, (_Float16)f0.z, (_Float16)f0.w };
        ua.q[1] = (v4h){ (_Float16)f1.x, (_Float16)f1.y, (_Float16)f1.z, (_Float16)f1.w };
        ua.q[2] = (v4h){ (_Float16)f2.x, (_Float16)f2.y, (_Float16)f2.z, (_Float16)f2.w };
        ua.q[3] = (v4h){ (_Float16)f3.x, (_Float16)f3.y, (_Float16)f3.z, (_Float16)f3.w };
        // V^T rows are valid B fragments (Bt[d][s] == Vt[d][s])
        v16h bv = load_b_frag_lds(Vbase + ks, SEQ, nt * 16 + ln, kh);
        o = wmma_f16(ua.v, bv, o);
    }
    #pragma unroll
    for (int r = 0; r < 8; ++r) {
        int srow = q0 + mt * 16 + r + ((lane >> 4) << 3);
        int col  = h * DK + nt * 16 + ln;
        ctx[(size_t)(b * SEQ + srow) * DMODEL + col] = (_Float16)o[r];
    }
}

// ---------------------------------------------------------------------------
// Kernel 3: FC + fused residual:  fc = ctx(f16) @ w_fc + q
// A tiles (already f16) staged via async global->LDS; 128x128 block tile.
// ---------------------------------------------------------------------------
__global__ __launch_bounds__(256) void fc_kernel(
    const _Float16* __restrict__ ctx, const float* __restrict__ wfc,
    const float* __restrict__ resid, float* __restrict__ fcout)
{
    extern __shared__ char smem3[];
    _Float16* As = (_Float16*)smem3;                          // 128 x LDA_STR
    _Float16* Bs = (_Float16*)(smem3 + 128 * LDA_STR * 2);    // 128 x LDB_STR
    const uint32_t dynb = __builtin_amdgcn_groupstaticsize();

    const int tid  = threadIdx.x;
    const int wave = tid >> 5;
    const int lane = tid & 31;
    const int M0 = blockIdx.y * 128;
    const int N0 = blockIdx.x * 128;
    const int wm = wave >> 1;
    const int wn = wave & 1;
    const int ln = lane & 15;
    const int kh  = (lane >> 4) * 8;
    const int kb2 = (lane >> 4) * 16;

    const int arow  = tid >> 1;            // A staging: 2 threads x 16 halves per row
    const int acolh = (tid & 1) * 16;
    const uint32_t aoff = dynb + (uint32_t)(arow * LDA_STR + acolh) * 2;

    v8f acc[2][4] = {};

    for (int k0 = 0; k0 < DMODEL; k0 += 32) {
        // A tile 128x32 f16 (8KB): raw async copy, 2 b128 per thread
        {
            const _Float16* gp = ctx + (size_t)(M0 + arow) * DMODEL + k0 + acolh;
            async_b128_to_lds(aoff,       gp);
            async_b128_to_lds(aoff + 16u, gp + 8);
        }
        // B tile 32x128 f32->f16 transposed -> Bs[n][k]
        #pragma unroll
        for (int i = 0; i < 4; ++i) {
            int idx4 = tid + i * 256;
            int kr = idx4 >> 5, c4 = idx4 & 31;
            float4 f = *(const float4*)&wfc[(size_t)(k0 + kr) * DMODEL + N0 + c4 * 4];
            Bs[(c4 * 4 + 0) * LDB_STR + kr] = (_Float16)f.x;
            Bs[(c4 * 4 + 1) * LDB_STR + kr] = (_Float16)f.y;
            Bs[(c4 * 4 + 2) * LDB_STR + kr] = (_Float16)f.z;
            Bs[(c4 * 4 + 3) * LDB_STR + kr] = (_Float16)f.w;
        }
        async_wait0();
        __syncthreads();

        v16h a0 = load_a_frag(As, LDA_STR, wm * 32 + ln,      kh);
        v16h a1 = load_a_frag(As, LDA_STR, wm * 32 + 16 + ln, kh);
        v16h bf[4];
        #pragma unroll
        for (int nt = 0; nt < 4; ++nt)
            bf[nt] = load_b_frag_lds(Bs, LDB_STR, wn * 64 + nt * 16 + ln, kh);
        #pragma unroll
        for (int nt = 0; nt < 4; ++nt) {
            acc[0][nt] = wmma_f16(a0, bf[nt], acc[0][nt]);
            acc[1][nt] = wmma_f16(a1, bf[nt], acc[1][nt]);
        }
        __syncthreads();
    }

    #pragma unroll
    for (int mt = 0; mt < 2; ++mt)
    #pragma unroll
    for (int nt = 0; nt < 4; ++nt)
    #pragma unroll
    for (int r = 0; r < 8; ++r) {
        int gm = M0 + wm * 32 + mt * 16 + r + ((lane >> 4) << 3);
        int gn = N0 + wn * 64 + nt * 16 + ln;
        size_t idx = (size_t)gm * DMODEL + gn;
        fcout[idx] = acc[mt][nt][r] + resid[idx];   // fused residual
    }
}

// ---------------------------------------------------------------------------
// Kernel 4: LayerNorm over rows of fc (residual already included)
// ---------------------------------------------------------------------------
__global__ __launch_bounds__(256) void ln_kernel(
    const float* __restrict__ x, const float* __restrict__ gamma,
    const float* __restrict__ beta, float* __restrict__ out)
{
    __shared__ float red[2][8];
    const int row  = blockIdx.x;
    const int tid  = threadIdx.x;
    const int wave = tid >> 5;
    const int lane = tid & 31;

    const float* xr = x + (size_t)row * DMODEL;
    float vals[4];
    float s = 0.0f, q = 0.0f;
    #pragma unroll
    for (int i = 0; i < 4; ++i) {
        float xx = xr[tid + i * 256];
        vals[i] = xx;
        s += xx;
        q += xx * xx;
    }
    #pragma unroll
    for (int off = 16; off > 0; off >>= 1) {
        s += __shfl_xor(s, off, 32);
        q += __shfl_xor(q, off, 32);
    }
    if (lane == 0) { red[0][wave] = s; red[1][wave] = q; }
    __syncthreads();

    float ts = 0.0f, tq = 0.0f;
    #pragma unroll
    for (int j = 0; j < 8; ++j) { ts += red[0][j]; tq += red[1][j]; }
    float mean = ts * (1.0f / DMODEL);
    float var  = tq * (1.0f / DMODEL) - mean * mean;
    float rstd = rsqrtf(var + 1e-5f);

    float* orow = out + (size_t)row * DMODEL;
    #pragma unroll
    for (int i = 0; i < 4; ++i) {
        int c = tid + i * 256;
        orow[c] = (vals[i] - mean) * rstd * gamma[c] + beta[c];
    }
}

// ---------------------------------------------------------------------------
// kernel_launch
// ---------------------------------------------------------------------------
extern "C" void kernel_launch(void* const* d_in, const int* in_sizes, int n_in,
                              void* d_out, int out_size, void* d_ws, size_t ws_size,
                              hipStream_t stream)
{
    const float* q     = (const float*)d_in[0];
    const float* k     = (const float*)d_in[1];
    const float* v     = (const float*)d_in[2];
    const float* wq    = (const float*)d_in[3];
    const float* wk    = (const float*)d_in[4];
    const float* wv    = (const float*)d_in[5];
    const float* wfc   = (const float*)d_in[6];
    const float* gamma = (const float*)d_in[7];
    const float* beta  = (const float*)d_in[8];

    float* out  = (float*)d_out;
    float* attn = out + (size_t)BATCH * SEQ * DMODEL;   // tuple: (out, attn)

    char* ws = (char*)d_ws;
    const size_t HELEMS = (size_t)BATCH * NHEAD * SEQ * DK;   // 4M halves each
    _Float16* Qh  = (_Float16*)ws;
    _Float16* Kh  = Qh + HELEMS;
    _Float16* Vt  = Kh + HELEMS;
    _Float16* ctx = Vt + HELEMS;                              // [B*S][1024] f16
    float*    fcb = (float*)(ctx + (size_t)BATCH * SEQ * DMODEL);

    const dim3 blk(256);

    // 1) QKV projections (z selects Q/K/V), 128x128 block tiles
    dim3 g1(DMODEL / 128, (BATCH * SEQ) / 128, 3);
    qkv_proj_kernel<<<g1, blk, 0, stream>>>(q, k, v, wq, wk, wv, Qh, Kh, Vt);

    // 2) Attention (262 KB dynamic LDS score tile — CDNA5 320KB/WG)
    const size_t smem2 = (size_t)32 * LDQ_STR * 2 + (size_t)32 * SSTR * 4;
    (void)hipFuncSetAttribute((const void*)attn_kernel,
                              hipFuncAttributeMaxDynamicSharedMemorySize, (int)smem2);
    dim3 g2(SEQ / 32, NHEAD, BATCH);
    attn_kernel<<<g2, blk, smem2, stream>>>(Qh, Kh, Vt, ctx, attn);

    // 3) FC + residual (async A staging), 128x128 block tiles
    const size_t smem3 = (size_t)128 * LDA_STR * 2 + (size_t)128 * LDB_STR * 2;
    dim3 g3(DMODEL / 128, (BATCH * SEQ) / 128, 1);
    fc_kernel<<<g3, blk, smem3, stream>>>(ctx, wfc, q, fcb);

    // 4) LayerNorm
    ln_kernel<<<dim3(BATCH * SEQ), blk, 0, stream>>>(fcb, gamma, beta, out);
}
